// MyVit_38182259262010
// MI455X (gfx1250) — matrix operand order repaired and checked by reading, not compiled, not run on hardware
//
#include <hip/hip_runtime.h>
#include <math.h>

// ---------------------------------------------------------------------------
// Types for CDNA5 WMMA (wave32, v_wmma_f32_16x16x32_f16)
// ---------------------------------------------------------------------------
typedef _Float16 half_t;
typedef __attribute__((ext_vector_type(16))) _Float16 v16h;
typedef __attribute__((ext_vector_type(8)))  _Float16 v8h;
typedef __attribute__((ext_vector_type(4)))  _Float16 v4h;
typedef __attribute__((ext_vector_type(8)))  float    v8f;

#define TILE 64      // output tile (M and N)
#define KB   64      // K chunk staged per barrier period (2 WMMA sub-steps)
#define LDSS 72      // padded LDS row stride in halves (144B = 36 banks)

enum { EPI_SCALE = 0, EPI_BIAS = 1, EPI_BIAS_GELU = 2, EPI_RES = 3, EPI_BIAS_RES = 4 };

// ---------------------------------------------------------------------------
// Generic batched GEMM:  C = epi(A[M,K] @ W[N,K]^T * scale + bias [+ res])
// TRANSB=true:  B given as [K,N] (row stride ldb), transposed while staging.
// BHALF=true:   B is pre-converted f16; full tiles staged with
//               GLOBAL_LOAD_ASYNC_TO_LDS_B128 (ASYNCcnt path).
// Batch z decomposes as (zo = z/IC, zi = z%IC); offsets = zo*so + zi*si.
// ---------------------------------------------------------------------------
template <int EPI, bool TRANSB, bool BHALF>
__global__ __launch_bounds__(128) void gemm_wmma(
    const float* __restrict__ A, long lda, long soA, long siA,
    const void* __restrict__ Bv, long ldb, long soB, long siB,
    const float* __restrict__ bias, long siBias,
    const float* __restrict__ res, long ldr, long soR, long siR,
    float* __restrict__ C, long ldc, long soC, long siC,
    int M, int N, int K, int IC, float scale)
{
    const int z  = blockIdx.z;
    const int zo = z / IC, zi = z % IC;
    A += (long)zo * soA + (long)zi * siA;
    const half_t* Bh = nullptr;
    const float*  Bf = nullptr;
    if (BHALF) Bh = (const half_t*)Bv + (long)zo * soB + (long)zi * siB;
    else       Bf = (const float*) Bv + (long)zo * soB + (long)zi * siB;
    C += (long)zo * soC + (long)zi * siC;
    if (EPI == EPI_RES || EPI == EPI_BIAS_RES) res += (long)zo * soR + (long)zi * siR;
    if (EPI == EPI_BIAS || EPI == EPI_BIAS_GELU || EPI == EPI_BIAS_RES) bias += (long)zi * siBias;

    const int tid  = threadIdx.x;
    const int wave = tid >> 5;
    const int lane = tid & 31;
    const int tM   = blockIdx.y * TILE;
    const int tN   = blockIdx.x * TILE;

    const bool fullM = (tM + TILE <= M);
    const bool fullN = (tN + TILE <= N);
    const bool vecA  = fullM && ((lda & 3) == 0);
    const bool vecBf = (!TRANSB) && (!BHALF) && fullN && ((ldb & 3) == 0);
    const bool vecBh = BHALF && fullN && ((ldb & 7) == 0);

    __shared__ half_t sA[TILE * LDSS];
    __shared__ half_t sB[TILE * LDSS];

    v8f acc[4];
#pragma unroll
    for (int t = 0; t < 4; ++t)
#pragma unroll
        for (int i = 0; i < 8; ++i) acc[t][i] = 0.0f;

    for (int k0 = 0; k0 < K; k0 += KB) {
        const bool fullK = (k0 + KB <= K);

        if (k0 + KB < K) {  // prefetch next K chunk (global_prefetch_b8)
            __builtin_prefetch(&A[(long)(tM + (tid >> 1)) * lda + k0 + KB + (tid & 1) * 32], 0, 3);
            if (!TRANSB && !BHALF)
                __builtin_prefetch(&Bf[(long)(tN + (tid >> 1)) * ldb + k0 + KB + (tid & 1) * 32], 0, 3);
        }

        // ---- stage A tile [64 rows][64 k] as f16 ----
        if (vecA && fullK) {
#pragma unroll
            for (int it = 0; it < 8; ++it) {
                int idx4 = (it * 128 + tid) << 2;         // 0..4095, step 4
                int r = idx4 >> 6, kk = idx4 & 63;
                const float4 v = *reinterpret_cast<const float4*>(&A[(long)(tM + r) * lda + k0 + kk]);
                v4h h4;
                h4[0] = (half_t)v.x; h4[1] = (half_t)v.y;
                h4[2] = (half_t)v.z; h4[3] = (half_t)v.w;
                *reinterpret_cast<v4h*>(&sA[r * LDSS + kk]) = h4;
            }
        } else {
            for (int idx = tid; idx < TILE * KB; idx += 128) {
                int r = idx >> 6, kk = idx & 63;
                int gm = tM + r, gk = k0 + kk;
                int gmc = gm < M ? gm : M - 1;
                int gkc = gk < K ? gk : K - 1;
                float v = A[(long)gmc * lda + gkc];       // clamped, always in-bounds
                v = (gm < M && gk < K) ? v : 0.0f;        // select, no branch
                sA[r * LDSS + kk] = (half_t)v;
            }
        }

        // ---- stage B tile as [64 cols][64 k] (W layout) ----
        if (BHALF) {
            if (vecBh && fullK) {
                // async DMA: 64 rows x 8 x 16B chunks, 4 per thread (ASYNCcnt)
#pragma unroll
                for (int it = 0; it < 4; ++it) {
                    int chunk = it * 128 + tid;           // 0..511
                    int r  = chunk >> 3;                  // row 0..63
                    int kk = (chunk & 7) << 3;            // half offset 0..56
                    const half_t* gp = &Bh[(long)(tN + r) * ldb + k0 + kk];
                    unsigned lp = (unsigned)(unsigned long long)&sB[r * LDSS + kk];
                    asm volatile("global_load_async_to_lds_b128 %0, %1, off"
                                 :: "v"(lp), "v"(gp) : "memory");
                }
                asm volatile("s_wait_asynccnt 0" ::: "memory");
            } else {
                for (int idx = tid; idx < TILE * KB; idx += 128) {
                    int r = idx >> 6, kk = idx & 63;
                    int gn = tN + r, gk = k0 + kk;
                    int gnc = gn < N ? gn : N - 1;
                    int gkc = gk < K ? gk : K - 1;
                    half_t v = Bh[(long)gnc * ldb + gkc];
                    sB[r * LDSS + kk] = (gn < N && gk < K) ? v : (half_t)0.0f;
                }
            }
        } else if (vecBf && fullK) {
#pragma unroll
            for (int it = 0; it < 8; ++it) {
                int idx4 = (it * 128 + tid) << 2;
                int r = idx4 >> 6, kk = idx4 & 63;
                const float4 v = *reinterpret_cast<const float4*>(&Bf[(long)(tN + r) * ldb + k0 + kk]);
                v4h h4;
                h4[0] = (half_t)v.x; h4[1] = (half_t)v.y;
                h4[2] = (half_t)v.z; h4[3] = (half_t)v.w;
                *reinterpret_cast<v4h*>(&sB[r * LDSS + kk]) = h4;
            }
        } else if (!TRANSB) {
            for (int idx = tid; idx < TILE * KB; idx += 128) {
                int r = idx >> 6, kk = idx & 63;
                int gn = tN + r, gk = k0 + kk;
                int gnc = gn < N ? gn : N - 1;
                int gkc = gk < K ? gk : K - 1;
                float v = Bf[(long)gnc * ldb + gkc];
                v = (gn < N && gk < K) ? v : 0.0f;
                sB[r * LDSS + kk] = (half_t)v;
            }
        } else {
            for (int idx = tid; idx < TILE * KB; idx += 128) {
                int c = idx & 63, kk = idx >> 6;          // coalesced along N
                int gn = tN + c, gk = k0 + kk;
                int gnc = gn < N ? gn : N - 1;
                int gkc = gk < K ? gk : K - 1;
                float v = Bf[(long)gkc * ldb + gnc];
                v = (gn < N && gk < K) ? v : 0.0f;
                sB[c * LDSS + kk] = (half_t)v;
            }
        }
        __syncthreads();

        // ---- 2 K sub-steps x 4 column tiles = 8 WMMAs per barrier period ----
#pragma unroll
        for (int ks = 0; ks < KB; ks += 32) {
            // A fragment: rows wave*16..+15; ISA 16-bit A 16x32 layout
            v16h af;
            {
                int r  = wave * 16 + (lane & 15);
                int kb = ks + (lane >> 4) * 8;            // lanes 16-31: K+8 / K+24
                v8h lo = *reinterpret_cast<const v8h*>(&sA[r * LDSS + kb]);
                v8h hi = *reinterpret_cast<const v8h*>(&sA[r * LDSS + kb + 16]);
                af = __builtin_shufflevector(lo, hi, 0,1,2,3,4,5,6,7,8,9,10,11,12,13,14,15);
            }
#pragma unroll
            for (int t = 0; t < 4; ++t) {
                int c   = t * 16 + (lane & 15);
                int kb2 = ks + (lane >> 4) * 8;           // lanes 16-31: K+8 / K+24
                v8h lo = *reinterpret_cast<const v8h*>(&sB[c * LDSS + kb2]);
                v8h hi = *reinterpret_cast<const v8h*>(&sB[c * LDSS + kb2 + 16]);
                v16h bf = __builtin_shufflevector(lo, hi, 0,1,2,3,4,5,6,7,8,9,10,11,12,13,14,15);
                acc[t] = __builtin_amdgcn_wmma_f32_16x16x32_f16(
                    false, af, false, bf, (short)0, acc[t], false, false);
            }
        }
        __syncthreads();
    }

    // ---- epilogue: C/D layout: VGPR r -> M = r + 8*(lane>>4), N = lane&15 ----
#pragma unroll
    for (int t = 0; t < 4; ++t) {
        int n = tN + t * 16 + (lane & 15);
#pragma unroll
        for (int r = 0; r < 8; ++r) {
            int m = tM + wave * 16 + r + ((lane >> 4) << 3);
            if (m < M && n < N) {
                float v = acc[t][r] * scale;
                if (EPI == EPI_BIAS || EPI == EPI_BIAS_GELU || EPI == EPI_BIAS_RES)
                    v += bias[n];
                if (EPI == EPI_BIAS_GELU)
                    v = 0.5f * v * (1.0f + erff(v * 0.70710678118654752f));
                if (EPI == EPI_RES || EPI == EPI_BIAS_RES)
                    v += res[(long)m * ldr + n];
                C[(long)m * ldc + n] = v;
            }
        }
    }
}

// ---------------------------------------------------------------------------
// f32 -> f16 weight conversion (vectorized, totals are multiples of 4)
// ---------------------------------------------------------------------------
__global__ __launch_bounds__(256) void f32_to_f16_kernel(
    const float* __restrict__ in, half_t* __restrict__ out, long total4)
{
    long i = (long)blockIdx.x * 256 + threadIdx.x;
    if (i >= total4) return;
    long i4 = i << 2;
    const float4 v = *reinterpret_cast<const float4*>(&in[i4]);
    v4h h4;
    h4[0] = (half_t)v.x; h4[1] = (half_t)v.y;
    h4[2] = (half_t)v.z; h4[3] = (half_t)v.w;
    *reinterpret_cast<v4h*>(&out[i4]) = h4;
}

// ---------------------------------------------------------------------------
// LayerNorm: one 256-thread block per row of length D
// ---------------------------------------------------------------------------
__global__ __launch_bounds__(256) void layernorm_kernel(
    const float* __restrict__ x, const float* __restrict__ w,
    const float* __restrict__ b, float* __restrict__ out, int D)
{
    const long row = blockIdx.x;
    const float* xr = x + row * D;
    float* orow = out + row * D;
    __shared__ float red[8];
    const int tid = threadIdx.x, lane = tid & 31, wv = tid >> 5;

    float s = 0.0f;
    for (int i = tid; i < D; i += 256) s += xr[i];
    for (int o = 16; o; o >>= 1) s += __shfl_xor(s, o, 32);
    if (lane == 0) red[wv] = s;
    __syncthreads();
    float mean = 0.0f;
    for (int i = 0; i < 8; ++i) mean += red[i];
    mean *= (1.0f / D);
    __syncthreads();

    float v = 0.0f;
    for (int i = tid; i < D; i += 256) { float d = xr[i] - mean; v += d * d; }
    for (int o = 16; o; o >>= 1) v += __shfl_xor(v, o, 32);
    if (lane == 0) red[wv] = v;
    __syncthreads();
    float var = 0.0f;
    for (int i = 0; i < 8; ++i) var += red[i];
    var *= (1.0f / D);
    float rstd = rsqrtf(var + 1e-5f);
    for (int i = tid; i < D; i += 256) orow[i] = (xr[i] - mean) * rstd * w[i] + b[i];
}

// ---------------------------------------------------------------------------
// Row softmax (in place): one 128-thread block per row
// ---------------------------------------------------------------------------
__global__ __launch_bounds__(128) void softmax_kernel(float* __restrict__ p, int cols, long ld)
{
    float* row = p + (long)blockIdx.x * ld;
    __shared__ float red[4];
    const int tid = threadIdx.x, lane = tid & 31, wv = tid >> 5;

    float mx = -INFINITY;
    for (int i = tid; i < cols; i += 128) mx = fmaxf(mx, row[i]);
    for (int o = 16; o; o >>= 1) mx = fmaxf(mx, __shfl_xor(mx, o, 32));
    if (lane == 0) red[wv] = mx;
    __syncthreads();
    mx = fmaxf(fmaxf(red[0], red[1]), fmaxf(red[2], red[3]));
    __syncthreads();

    float s = 0.0f;
    for (int i = tid; i < cols; i += 128) { float e = expf(row[i] - mx); row[i] = e; s += e; }
    for (int o = 16; o; o >>= 1) s += __shfl_xor(s, o, 32);
    if (lane == 0) red[wv] = s;
    __syncthreads();
    s = red[0] + red[1] + red[2] + red[3];
    float inv = 1.0f / s;
    for (int i = tid; i < cols; i += 128) row[i] *= inv;
}

// ---------------------------------------------------------------------------
// Patchify: images[B,3,224,224] -> patches[B*196, 768] (C,ph,pw flatten order)
// ---------------------------------------------------------------------------
__global__ __launch_bounds__(256) void patchify_kernel(
    const float* __restrict__ img, float* __restrict__ patches, int total)
{
    int idx = blockIdx.x * 256 + threadIdx.x;
    if (idx >= total) return;
    int k  = idx % 768;
    int mp = idx / 768;
    int b = mp / 196, p = mp % 196;
    int c = k >> 8, r = (k >> 4) & 15, cc = k & 15;
    int py = p / 14, px = p % 14;
    patches[idx] = img[((long)(b * 3 + c) * 224 + py * 16 + r) * 224 + px * 16 + cc];
}

// ---------------------------------------------------------------------------
// Assemble x = [cls | tokens] + pos_embed
// ---------------------------------------------------------------------------
__global__ __launch_bounds__(256) void assemble_kernel(
    const float* __restrict__ tokens, const float* __restrict__ cls,
    const float* __restrict__ pos, float* __restrict__ x, int total)
{
    int idx = blockIdx.x * 256 + threadIdx.x;
    if (idx >= total) return;
    int d  = idx % 768;
    int bs = idx / 768;
    int s  = bs % 197;
    int b  = bs / 197;
    float v = (s == 0) ? cls[d] : tokens[((long)b * 196 + (s - 1)) * 768 + d];
    x[idx] = v + pos[(long)s * 768 + d];
}

// ---------------------------------------------------------------------------
// Host orchestration
// ---------------------------------------------------------------------------
static inline dim3 ggrid(int M, int N, int Z) {
    return dim3((unsigned)((N + 63) / 64), (unsigned)((M + 63) / 64), (unsigned)Z);
}

extern "C" void kernel_launch(void* const* d_in, const int* in_sizes, int n_in,
                              void* d_out, int out_size, void* d_ws, size_t ws_size,
                              hipStream_t stream) {
    (void)in_sizes; (void)n_in; (void)out_size; (void)ws_size;
    const float* images  = (const float*)d_in[0];
    const float* W_map   = (const float*)d_in[1];
    const float* b_map   = (const float*)d_in[2];
    const float* cls_tok = (const float*)d_in[3];
    const float* pos     = (const float*)d_in[4];
    const float* ln1_w   = (const float*)d_in[5];
    const float* ln1_b   = (const float*)d_in[6];
    const float* Wq      = (const float*)d_in[7];
    const float* bq      = (const float*)d_in[8];
    const float* Wk      = (const float*)d_in[9];
    const float* bk      = (const float*)d_in[10];
    const float* Wv      = (const float*)d_in[11];
    const float* bv      = (const float*)d_in[12];
    const float* ln2_w   = (const float*)d_in[13];
    const float* ln2_b   = (const float*)d_in[14];
    const float* W1      = (const float*)d_in[15];
    const float* b1      = (const float*)d_in[16];
    const float* W2      = (const float*)d_in[17];
    const float* b2      = (const float*)d_in[18];
    const float* W_head  = (const float*)d_in[19];
    const float* b_head  = (const float*)d_in[20];
    float* out = (float*)d_out;

    const int Dm = 768, Fm = 3072, Sq = 197, Bc = 16, NH = 12;
    const int MS = Bc * Sq;          // 3152
    const int MT = Bc * 196;         // 3136
    const long XB = (long)Sq * Dm;   // 151296  per-batch x stride
    const long SS = (long)Sq * Sq;   // 38809   per-(b,h) score stride

    float* ws = (float*)d_ws;
    float* mbuf    = ws;                              // 3152*3072   (layer loop)
    float* patches = ws;                              // aliased (pre-loop only)
    float* tokens  = ws + 2408448;                    // aliased (pre-loop only)
    float* x   = ws + 9682944;                        // 3152*768
    float* h   = x  + 2420736;                        // 3152*768
    float* qb  = h  + 2420736;                        // [B,S,NH*DH]
    float* kb  = qb + 2420736;
    float* vb  = kb + 2420736;
    float* sc  = vb + 2420736;                        // 192*197*197

    // f16 weight arena (after the f32 region: 29,237,952 floats)
    half_t* hW_map = (half_t*)(ws + 29237952);        // 589,824
    half_t* hWq    = hW_map + 589824;                 // 12*12*64*64
    half_t* hWk    = hWq    + 589824;
    half_t* hWv    = hWk    + 589824;
    half_t* hW1    = hWv    + 589824;                 // 12*3072*768
    half_t* hW2    = hW1    + 28311552;
    half_t* hWhd   = hW2    + 28311552;               // 1000*768

    // ---- one-time weight conversion to f16 ----
    f32_to_f16_kernel<<<(589824 / 4 + 255) / 256, 256, 0, stream>>>(W_map, hW_map, 589824 / 4);
    f32_to_f16_kernel<<<(589824 / 4 + 255) / 256, 256, 0, stream>>>(Wq, hWq, 589824 / 4);
    f32_to_f16_kernel<<<(589824 / 4 + 255) / 256, 256, 0, stream>>>(Wk, hWk, 589824 / 4);
    f32_to_f16_kernel<<<(589824 / 4 + 255) / 256, 256, 0, stream>>>(Wv, hWv, 589824 / 4);
    f32_to_f16_kernel<<<(28311552 / 4 + 255) / 256, 256, 0, stream>>>(W1, hW1, 28311552 / 4);
    f32_to_f16_kernel<<<(28311552 / 4 + 255) / 256, 256, 0, stream>>>(W2, hW2, 28311552 / 4);
    f32_to_f16_kernel<<<(768000 / 4 + 255) / 256, 256, 0, stream>>>(W_head, hWhd, 768000 / 4);

    // ---- patch embedding ----
    patchify_kernel<<<(MT * Dm + 255) / 256, 256, 0, stream>>>(images, patches, MT * Dm);
    gemm_wmma<EPI_BIAS, false, true><<<ggrid(MT, Dm, 1), 128, 0, stream>>>(
        patches, Dm, 0, 0,  hW_map, Dm, 0, 0,  b_map, 0,
        nullptr, 0, 0, 0,  tokens, Dm, 0, 0,  MT, Dm, Dm, 1, 1.0f);
    assemble_kernel<<<(MS * Dm + 255) / 256, 256, 0, stream>>>(tokens, cls_tok, pos, x, MS * Dm);

    // ---- transformer blocks ----
    for (int l = 0; l < 12; ++l) {
        const half_t* wqL = hWq + (long)l * NH * 64 * 64;
        const half_t* wkL = hWk + (long)l * NH * 64 * 64;
        const half_t* wvL = hWv + (long)l * NH * 64 * 64;
        const float*  bqL = bq + (long)l * Dm;
        const float*  bkL = bk + (long)l * Dm;
        const float*  bvL = bv + (long)l * Dm;

        layernorm_kernel<<<MS, 256, 0, stream>>>(x, ln1_w + l * Dm, ln1_b + l * Dm, h, Dm);

        // per-head QKV projections: 12 batched [MS,64]x[64,64]^T
        gemm_wmma<EPI_BIAS, false, true><<<ggrid(MS, 64, NH), 128, 0, stream>>>(
            h, Dm, 0, 64,  wqL, 64, 0, 4096,  bqL, 64,
            nullptr, 0, 0, 0,  qb, Dm, 0, 64,  MS, 64, 64, NH, 1.0f);
        gemm_wmma<EPI_BIAS, false, true><<<ggrid(MS, 64, NH), 128, 0, stream>>>(
            h, Dm, 0, 64,  wkL, 64, 0, 4096,  bkL, 64,
            nullptr, 0, 0, 0,  kb, Dm, 0, 64,  MS, 64, 64, NH, 1.0f);
        gemm_wmma<EPI_BIAS, false, true><<<ggrid(MS, 64, NH), 128, 0, stream>>>(
            h, Dm, 0, 64,  wvL, 64, 0, 4096,  bvL, 64,
            nullptr, 0, 0, 0,  vb, Dm, 0, 64,  MS, 64, 64, NH, 1.0f);

        // scores = q @ k^T / 8 over 192 (b,h) batches (B = f32 activations)
        gemm_wmma<EPI_SCALE, false, false><<<ggrid(Sq, Sq, Bc * NH), 128, 0, stream>>>(
            qb, Dm, XB, 64,  kb, Dm, XB, 64,  nullptr, 0,
            nullptr, 0, 0, 0,  sc, Sq, NH * SS, SS,  Sq, Sq, 64, NH, 0.125f);

        softmax_kernel<<<Bc * NH * Sq, 128, 0, stream>>>(sc, Sq, Sq);

        // x += attn @ v   (B supplied as [K=197, N=64] -> TRANSB staging)
        gemm_wmma<EPI_RES, true, false><<<ggrid(Sq, 64, Bc * NH), 128, 0, stream>>>(
            sc, Sq, NH * SS, SS,  vb, Dm, XB, 64,  nullptr, 0,
            x, Dm, XB, 64,  x, Dm, XB, 64,  Sq, 64, Sq, NH, 1.0f);

        layernorm_kernel<<<MS, 256, 0, stream>>>(x, ln2_w + l * Dm, ln2_b + l * Dm, h, Dm);

        // MLP
        gemm_wmma<EPI_BIAS_GELU, false, true><<<ggrid(MS, Fm, 1), 128, 0, stream>>>(
            h, Dm, 0, 0,  hW1 + (long)l * Fm * Dm, Dm, 0, 0,  b1 + (long)l * Fm, 0,
            nullptr, 0, 0, 0,  mbuf, Fm, 0, 0,  MS, Fm, Dm, 1, 1.0f);
        gemm_wmma<EPI_BIAS_RES, false, true><<<ggrid(MS, Dm, 1), 128, 0, stream>>>(
            mbuf, Fm, 0, 0,  hW2 + (long)l * Dm * Fm, Fm, 0, 0,  b2 + (long)l * Dm, 0,
            x, Dm, 0, 0,  x, Dm, 0, 0,  MS, Dm, Fm, 1, 1.0f);
    }

    // ---- classifier head on cls tokens (row stride = per-batch x stride) ----
    gemm_wmma<EPI_BIAS, false, true><<<ggrid(Bc, 1000, 1), 128, 0, stream>>>(
        x, XB, 0, 0,  hWhd, Dm, 0, 0,  b_head, 0,
        nullptr, 0, 0, 0,  out, 1000, 0, 0,  Bc, 1000, Dm, 1, 1.0f);
    softmax_kernel<<<Bc, 128, 0, stream>>>(out, 1000, 1000);
}
